// RCANE_59682865545580
// MI455X (gfx1250) — matrix operand order repaired and checked by reading, not compiled, not run on hardware
//
#include <hip/hip_runtime.h>
#include <hip/hip_bf16.h>
#include <math.h>
#include <stdint.h>

// ---------------------------------------------------------------------------
// Problem constants (from reference): B=4, N=10000, E=640000, IN=128,
// HID=OUT=64, H=2 heads.  Nn = 40000 nodes.
// ---------------------------------------------------------------------------
#define NN_NODES 40000
#define N_EDGES  640000
#define N_HEADS  2

typedef __attribute__((ext_vector_type(16))) __bf16 v16bf;
typedef __attribute__((ext_vector_type(8)))  __bf16 v8bf;
typedef __attribute__((ext_vector_type(8)))  float  v8f;
typedef __attribute__((ext_vector_type(4)))  unsigned int v4u;
typedef __attribute__((ext_vector_type(4)))  int v4i;
typedef __attribute__((ext_vector_type(8)))  int v8i;

// float -> bf16, round-to-nearest-even
__device__ inline __bf16 f2bf(float f) {
    unsigned u = __float_as_uint(f);
    u += 0x7FFFu + ((u >> 16) & 1u);
    unsigned short hs = (unsigned short)(u >> 16);
    return __builtin_bit_cast(__bf16, hs);
}

// float atomic max via signed/unsigned integer reinterpretation trick
__device__ inline void atomicMaxF(float* addr, float v) {
    if (v >= 0.0f) {
        atomicMax((int*)addr, __float_as_int(v));
    } else {
        atomicMin((unsigned int*)addr, (unsigned int)__float_as_int(v));
    }
}

__device__ inline unsigned rfl(unsigned x) {
    return (unsigned)__builtin_amdgcn_readfirstlane((int)x);
}

// ---------------------------------------------------------------------------
// TDM 2D tile load: global fp32 tensor -> LDS, via Tensor Data Mover.
// D# layout per CDNA5 ISA 8.3/8.4: group0 = {count/type/addrs}, group1 =
// {data_size, tensor dims, tile dims, strides}.  dims/strides in elements
// (data_size = 4 bytes).  OOB rows (y >= dim1) read as zero -> free M-padding.
// Issue from one wave; completion via s_wait_tensorcnt.
// ---------------------------------------------------------------------------
__device__ inline void tdm_load_2d(const void* gaddr, unsigned lds_off,
                                   unsigned dim0, unsigned dim1,
                                   unsigned tile0, unsigned tile1,
                                   unsigned stride0)
{
    unsigned long long ga = (unsigned long long)(uintptr_t)gaddr;
    v4u g0;
    g0[0] = rfl(1u);                                   // count=1 (valid user D#)
    g0[1] = rfl(lds_off);                              // lds_addr (bytes)
    g0[2] = rfl((unsigned)(ga & 0xFFFFFFFFu));         // global_addr[31:0]
    g0[3] = rfl((unsigned)((ga >> 32) & 0x01FFFFFFu)   // global_addr[56:32]
                | (2u << 30));                         // type=2 ("image")

    v8i g1;
    g1[0] = (int)rfl(2u << 16);                        // data_size=2 -> 4 bytes
    g1[1] = (int)rfl((dim0 & 0xFFFFu) << 16);          // tensor_dim0[15:0] @bit48
    g1[2] = (int)rfl(((dim0 >> 16) & 0xFFFFu)          // tensor_dim0[31:16]
                     | ((dim1 & 0xFFFFu) << 16));      // tensor_dim1[15:0]
    g1[3] = (int)rfl(((dim1 >> 16) & 0xFFFFu)          // tensor_dim1[31:16]
                     | ((tile0 & 0xFFFFu) << 16));     // tile_dim0
    g1[4] = (int)rfl(tile1 & 0xFFFFu);                 // tile_dim1 (tile_dim2=0)
    g1[5] = (int)rfl(stride0);                         // tensor_dim0_stride[31:0]
    g1[6] = (int)rfl(0u);                              // stride0 hi / dim1_stride lo
    g1[7] = (int)rfl(0u);                              // dim1_stride hi

    v4i gz = {0, 0, 0, 0};                             // groups 2/3 unused (<=2D)
#if defined(__clang_major__) && (__clang_major__ >= 23)
    v8i gz8 = {0, 0, 0, 0, 0, 0, 0, 0};
    __builtin_amdgcn_tensor_load_to_lds(g0, g1, gz, gz, gz8, 0);
#else
    __builtin_amdgcn_tensor_load_to_lds(g0, g1, gz, gz, 0);
#endif
}

// ---------------------------------------------------------------------------
// WMMA bf16 GEMM:  Y[M,N] = X[M,K] @ W[K,N] + bias[N]   (fp32 in/out)
// 256 threads = 8 waves; block tile = 128 rows x N cols.
// Stage fp32 tiles with the Tensor Data Mover, convert to bf16 in LDS
// (X row-major, W transposed [N][K]), then WMMA from LDS fragments.
// Fragment layouts per CDNA5 ISA 7.12.2 (16-bit A 16x32, B 32x16, f32 C/D).
// ---------------------------------------------------------------------------
template<int K, int N>
__global__ __launch_bounds__(256)
void gemm_bias_wmma(const float* __restrict__ X, const float* __restrict__ W,
                    const float* __restrict__ bias, float* __restrict__ Y,
                    int M)
{
    constexpr int BM = 128;
    constexpr int NT = N / 16;   // 16-wide output tiles per wave
    constexpr int KC = K / 32;   // 32-deep K chunks

    __shared__ float  Xraw[BM * K];  // TDM staging, fp32
    __shared__ float  Wraw[K * N];   // TDM staging, fp32
    __shared__ __bf16 Xs[BM * K];    // [row][k] bf16
    __shared__ __bf16 Wt[N * K];     // transposed [n][k] bf16

    const int tid      = threadIdx.x;
    const int row_base = blockIdx.x * BM;

    // ---- async DMA both tiles into LDS (wave 0 issues; TENSORcnt tracks) ----
    if (tid < 32) {
        // X tile: 2D, rows clamped by tensor_dim1 = M - row_base (OOB -> 0)
        tdm_load_2d(X + (long)row_base * K, (unsigned)(uintptr_t)&Xraw[0],
                    (unsigned)K, (unsigned)(M - row_base),
                    (unsigned)K, (unsigned)BM, (unsigned)K);
        // W: flat 1D tile of K*N elements
        tdm_load_2d(W, (unsigned)(uintptr_t)&Wraw[0],
                    (unsigned)(K * N), 1u, (unsigned)(K * N), 1u,
                    (unsigned)(K * N));
        __builtin_amdgcn_s_wait_tensorcnt(0);
    }
    __syncthreads();

    // ---- convert fp32 -> bf16 (and transpose W) ----
    for (int idx = tid; idx < BM * K; idx += 256)
        Xs[idx] = f2bf(Xraw[idx]);
    for (int idx = tid; idx < K * N; idx += 256) {
        const int k = idx / N;
        const int n = idx % N;
        Wt[n * K + k] = f2bf(Wraw[idx]);
    }
    __syncthreads();

    const int wave = tid >> 5;
    const int lane = tid & 31;
    const int half = lane >> 4;   // 0: lanes 0-15, 1: lanes 16-31
    const int lm   = lane & 15;

    // ---- A fragments for this wave's 16 rows (all K chunks) ----
    // 16-bit A 16x32 layout: lanes 0-15 hold K = [c..c+7] and [c+16..c+23],
    // lanes 16-31 hold K = [c+8..c+15] and [c+24..c+31].
    v16bf afrag[KC];
    const __bf16* xrow = &Xs[(wave * 16 + lm) * K];
#pragma unroll
    for (int kc = 0; kc < KC; ++kc) {
        v8bf lo = *(const v8bf*)(xrow + kc * 32 + half * 8);
        v8bf hi = *(const v8bf*)(xrow + kc * 32 + 16 + half * 8);
        afrag[kc] = __builtin_shufflevector(lo, hi,
            0, 1, 2, 3, 4, 5, 6, 7, 8, 9, 10, 11, 12, 13, 14, 15);
    }

    const bool full_tile = (row_base + BM) <= M;

#pragma unroll
    for (int nt = 0; nt < NT; ++nt) {
        v8f acc = {};
        // 16-bit B 32x16 layout: lane (&15) = column; lanes 0-15 hold
        // K = [c..c+7] and [c+16..c+23], lanes 16-31 hold K = [c+8..c+15] and [c+24..c+31].
        const __bf16* wcol = &Wt[(nt * 16 + lm) * K];
#pragma unroll
        for (int kc = 0; kc < KC; ++kc) {
            v8bf b0 = *(const v8bf*)(wcol + kc * 32 + half * 8);
            v8bf b1 = *(const v8bf*)(wcol + kc * 32 + 16 + half * 8);
            v16bf bfrag = __builtin_shufflevector(b0, b1,
                0, 1, 2, 3, 4, 5, 6, 7, 8, 9, 10, 11, 12, 13, 14, 15);
            acc = __builtin_amdgcn_wmma_f32_16x16x32_bf16(
                false, afrag[kc], false, bfrag, (short)0, acc, false, false);
        }
        const float bv = bias[nt * 16 + lm];
        // f32 C/D layout: VGPR r -> row (r + 8*half), col = lane&15
        float* yb = Y + (long)(row_base + wave * 16 + half * 8) * N + nt * 16 + lm;
        if (full_tile) {
#pragma unroll
            for (int r = 0; r < 8; ++r)
                yb[(long)r * N] = acc[r] + bv;
        } else {
#pragma unroll
            for (int r = 0; r < 8; ++r) {
                const int row = row_base + wave * 16 + r + half * 8;
                if (row < M)
                    yb[(long)r * N] = acc[r] + bv;
            }
        }
    }
}

// ---------------------------------------------------------------------------
// fill buffer with a constant
// ---------------------------------------------------------------------------
__global__ void fill_kernel(float* __restrict__ p, float v, long n)
{
    long i = (long)blockIdx.x * blockDim.x + threadIdx.x;
    if (i < n) p[i] = v;
}

// ---------------------------------------------------------------------------
// Edge logits: alpha[e,h] = (q[dst,h,:] . k[src,h,:]) * scale, plus
// atomic segment-max into m[dst,h].  One wave32 per (edge, head); each lane
// covers 2 of the 64 channels, reduced with shfl_xor.
// ---------------------------------------------------------------------------
template<int C>
__global__ __launch_bounds__(256)
void edge_logits(const long long* __restrict__ ei,
                 const float* __restrict__ Q, const float* __restrict__ Kf,
                 float* __restrict__ alpha, float* __restrict__ m,
                 int nE, int Hh, float scale)
{
    const int w    = blockIdx.x * (blockDim.x >> 5) + (threadIdx.x >> 5);
    const int lane = threadIdx.x & 31;
    if (w >= nE * Hh) return;
    const int e = w / Hh;
    const int h = w % Hh;
    const long src = ei[e];
    const long dst = ei[nE + e];

    const float* q = Q  + ((long)dst * Hh + h) * C;
    const float* k = Kf + ((long)src * Hh + h) * C;

    const float2 q2 = *(const float2*)(q + lane * 2);
    const float2 k2 = *(const float2*)(k + lane * 2);
    float p = q2.x * k2.x + q2.y * k2.y;
#pragma unroll
    for (int off = 16; off > 0; off >>= 1)
        p += __shfl_xor(p, off, 32);
    p *= scale;
    if (lane == 0) {
        alpha[(long)e * Hh + h] = p;
        atomicMaxF(&m[(long)dst * Hh + h], p);
    }
}

// ---------------------------------------------------------------------------
// Edge scatter: e = exp(alpha - m[dst]); denom[dst,h] += e;
// msg[dst,h,:] += e * v[src,h,:].  One wave32 per (edge, head).
// ---------------------------------------------------------------------------
template<int C>
__global__ __launch_bounds__(256)
void edge_scatter(const long long* __restrict__ ei,
                  const float* __restrict__ alpha, const float* __restrict__ m,
                  const float* __restrict__ V,
                  float* __restrict__ denom, float* __restrict__ msg,
                  int nE, int Hh)
{
    const int w    = blockIdx.x * (blockDim.x >> 5) + (threadIdx.x >> 5);
    const int lane = threadIdx.x & 31;
    if (w >= nE * Hh) return;
    const int e = w / Hh;
    const int h = w % Hh;
    const long src = ei[e];
    const long dst = ei[nE + e];

    const float a  = alpha[(long)e * Hh + h];
    const float mm = m[(long)dst * Hh + h];
    const float ee = __expf(a - mm);

    if (lane == 0) atomicAdd(&denom[(long)dst * Hh + h], ee);

    const float* v  = V   + ((long)src * Hh + h) * C;
    float*       mg = msg + ((long)dst * Hh + h) * C;
    const float2 v2 = *(const float2*)(v + lane * 2);
    atomicAdd(&mg[lane * 2 + 0], ee * v2.x);
    atomicAdd(&mg[lane * 2 + 1], ee * v2.y);
}

// ---------------------------------------------------------------------------
// Node finalize: out[n,c] = mean_h( msg[n,h,c] / max(denom[n,h],1e-16) )
//                + skip[n,c]   (+ leaky_relu(0.1) for layer 1)
// ---------------------------------------------------------------------------
template<int C, bool LRELU>
__global__ void node_finalize(const float* __restrict__ msg,
                              const float* __restrict__ denom,
                              const float* __restrict__ skip,
                              float* __restrict__ out, int Nn, int Hh)
{
    const long i = (long)blockIdx.x * blockDim.x + threadIdx.x;
    if (i >= (long)Nn * C) return;
    const int n = (int)(i / C);
    const int c = (int)(i % C);
    float acc = 0.0f;
    for (int h = 0; h < Hh; ++h) {
        const float d = denom[(long)n * Hh + h];
        acc += msg[((long)n * Hh + h) * C + c] / fmaxf(d, 1e-16f);
    }
    acc = acc / (float)Hh + skip[i];
    if (LRELU) acc = (acc > 0.0f) ? acc : 0.1f * acc;
    out[i] = acc;
}

// ---------------------------------------------------------------------------
// Launcher
// ---------------------------------------------------------------------------
extern "C" void kernel_launch(void* const* d_in, const int* in_sizes, int n_in,
                              void* d_out, int out_size, void* d_ws, size_t ws_size,
                              hipStream_t stream)
{
    (void)in_sizes; (void)n_in; (void)out_size; (void)ws_size;

    const float*     x  = (const float*)d_in[0];
    const long long* ei = (const long long*)d_in[1];
    const float *Wq1 = (const float*)d_in[2],  *bq1 = (const float*)d_in[3];
    const float *Wk1 = (const float*)d_in[4],  *bk1 = (const float*)d_in[5];
    const float *Wv1 = (const float*)d_in[6],  *bv1 = (const float*)d_in[7];
    const float *Ws1 = (const float*)d_in[8],  *bs1 = (const float*)d_in[9];
    const float *Wq2 = (const float*)d_in[10], *bq2 = (const float*)d_in[11];
    const float *Wk2 = (const float*)d_in[12], *bk2 = (const float*)d_in[13];
    const float *Wv2 = (const float*)d_in[14], *bv2 = (const float*)d_in[15];
    const float *Ws2 = (const float*)d_in[16], *bs2 = (const float*)d_in[17];
    float* out = (float*)d_out;

    const int Nn = NN_NODES;
    const int nE = N_EDGES;
    const int Hh = N_HEADS;

    // workspace layout (floats)
    float* ws    = (float*)d_ws;
    float* Qb    = ws;                          // Nn*128
    float* Kb    = Qb    + (long)Nn * 128;      // Nn*128
    float* Vb    = Kb    + (long)Nn * 128;      // Nn*128
    float* Sk    = Vb    + (long)Nn * 128;      // Nn*64
    float* alpha = Sk    + (long)Nn * 64;       // E*H
    float* mbuf  = alpha + (long)nE * Hh;       // Nn*H
    float* denom = mbuf  + (long)Nn * Hh;       // Nn*H
    float* msg   = denom + (long)Nn * Hh;       // Nn*H*64
    float* h1    = msg   + (long)Nn * Hh * 64;  // Nn*64

    const dim3 blk(256);
    const int  gm    = (Nn + 127) / 128;                 // gemm blocks
    const int  gedge = (nE * Hh + 7) / 8;                // 8 waves / block
    const int  gnode = ((long)Nn * 64 + 255) / 256;
    const float scale = 0.125f;                          // 1/sqrt(64)

    // ================= layer 1 (IN=128 -> HID=64, H=2) =================
    gemm_bias_wmma<128, 128><<<gm, blk, 0, stream>>>(x, Wq1, bq1, Qb, Nn);
    gemm_bias_wmma<128, 128><<<gm, blk, 0, stream>>>(x, Wk1, bk1, Kb, Nn);
    gemm_bias_wmma<128, 128><<<gm, blk, 0, stream>>>(x, Wv1, bv1, Vb, Nn);
    gemm_bias_wmma<128, 64><<<gm, blk, 0, stream>>>(x, Ws1, bs1, Sk, Nn);

    fill_kernel<<<((long)Nn * Hh + 255) / 256, blk, 0, stream>>>(mbuf, -3.0e38f, (long)Nn * Hh);
    fill_kernel<<<((long)Nn * Hh + 255) / 256, blk, 0, stream>>>(denom, 0.0f, (long)Nn * Hh);
    fill_kernel<<<((long)Nn * Hh * 64 + 255) / 256, blk, 0, stream>>>(msg, 0.0f, (long)Nn * Hh * 64);

    edge_logits<64><<<gedge, blk, 0, stream>>>(ei, Qb, Kb, alpha, mbuf, nE, Hh, scale);
    edge_scatter<64><<<gedge, blk, 0, stream>>>(ei, alpha, mbuf, Vb, denom, msg, nE, Hh);
    node_finalize<64, true><<<gnode, blk, 0, stream>>>(msg, denom, Sk, h1, Nn, Hh);

    // ================= layer 2 (HID=64 -> OUT=64, H=2) =================
    gemm_bias_wmma<64, 128><<<gm, blk, 0, stream>>>(h1, Wq2, bq2, Qb, Nn);
    gemm_bias_wmma<64, 128><<<gm, blk, 0, stream>>>(h1, Wk2, bk2, Kb, Nn);
    gemm_bias_wmma<64, 128><<<gm, blk, 0, stream>>>(h1, Wv2, bv2, Vb, Nn);
    gemm_bias_wmma<64, 64><<<gm, blk, 0, stream>>>(h1, Ws2, bs2, Sk, Nn);

    fill_kernel<<<((long)Nn * Hh + 255) / 256, blk, 0, stream>>>(mbuf, -3.0e38f, (long)Nn * Hh);
    fill_kernel<<<((long)Nn * Hh + 255) / 256, blk, 0, stream>>>(denom, 0.0f, (long)Nn * Hh);
    fill_kernel<<<((long)Nn * Hh * 64 + 255) / 256, blk, 0, stream>>>(msg, 0.0f, (long)Nn * Hh * 64);

    edge_logits<64><<<gedge, blk, 0, stream>>>(ei, Qb, Kb, alpha, mbuf, nE, Hh, scale);
    edge_scatter<64><<<gedge, blk, 0, stream>>>(ei, alpha, mbuf, Vb, denom, msg, nE, Hh);
    node_finalize<64, false><<<gnode, blk, 0, stream>>>(msg, denom, Sk, out, Nn, Hh);
}
